// SMoE_343597384323
// MI455X (gfx1250) — compile-verified, matrix operand order checked by reading, not run on hardware
//
#include <hip/hip_runtime.h>
#include <cstdint>
#include <cstddef>

// Problem sizes (match reference)
#define BB 4
#define SS 2048
#define DD 1024
#define FF 2048
#define EE 8
#define TT (BB * SS) // 8192 tokens

typedef _Float16 half_t;
typedef _Float16 halfx16 __attribute__((ext_vector_type(16)));
typedef float floatx8 __attribute__((ext_vector_type(8)));

#if __has_builtin(__builtin_amdgcn_global_load_async_to_lds_b128) && \
    __has_builtin(__builtin_amdgcn_s_wait_asynccnt)
#define USE_ASYNC_LDS 1
// Exact pointee type from the builtin's prototype (per hipcc diagnostic):
typedef int v4i __attribute__((vector_size(16)));
typedef __attribute__((address_space(1))) v4i gv4i_t; // global (prints as __device__)
typedef __attribute__((address_space(3))) v4i lv4i_t; // LDS
#endif

union Frag {
    uint4 u[2];
    halfx16 v;
};

// Load a 16-element f16 WMMA fragment from two non-contiguous 16B chunks in LDS.
static __device__ __forceinline__ halfx16 ld_frag(const half_t* lo, const half_t* hi) {
    Frag f;
    f.u[0] = *(const uint4*)lo;
    f.u[1] = *(const uint4*)hi;
    return f.v;
}

// ---------------------------------------------------------------------------
// Kernel 1: RMSNorm + gating + routing. One block (256 threads) per token.
// ---------------------------------------------------------------------------
__global__ __launch_bounds__(256) void rmsnorm_gate_kernel(
    const float* __restrict__ x, const float* __restrict__ rms_w,
    const float* __restrict__ Wg, const float* __restrict__ bg,
    half_t* __restrict__ xn, int* __restrict__ counts,
    int* __restrict__ lists, float* __restrict__ gatew,
    float* __restrict__ wout, float* __restrict__ out)
{
    const int t = blockIdx.x;
    const int tid = threadIdx.x;
    const float* xr = x + (size_t)t * DD;

    float v[4];
    float ss = 0.f;
#pragma unroll
    for (int i = 0; i < 4; ++i) {
        v[i] = xr[tid + i * 256];
        ss += v[i] * v[i];
    }

    __shared__ float red[256];
    red[tid] = ss;
    __syncthreads();
#pragma unroll
    for (int s = 128; s > 0; s >>= 1) {
        if (tid < s) red[tid] += red[tid + s];
        __syncthreads();
    }
    const float scale = rsqrtf(red[0] * (1.0f / DD) + 1.1920929e-7f); // fp32 eps

    __shared__ float sLog[EE];
    if (tid < EE) sLog[tid] = bg[tid];
    __syncthreads();

    float part[EE];
#pragma unroll
    for (int e = 0; e < EE; ++e) part[e] = 0.f;

#pragma unroll
    for (int i = 0; i < 4; ++i) {
        const int d = tid + i * 256;
        const float nv = v[i] * scale * rms_w[d];
        xn[(size_t)t * DD + d] = (half_t)nv;
#pragma unroll
        for (int e = 0; e < EE; ++e) part[e] += nv * Wg[d * EE + e];
    }
#pragma unroll
    for (int e = 0; e < EE; ++e) atomicAdd(&sLog[e], part[e]);

    // zero output row so kernel 2 can atomically accumulate
    float* orow = out + (size_t)t * DD;
#pragma unroll
    for (int i = 0; i < 4; ++i) orow[tid + i * 256] = 0.f;

    __syncthreads();
    if (tid == 0) {
        float l[EE];
#pragma unroll
        for (int e = 0; e < EE; ++e) l[e] = sLog[e];
        int i0 = 0;
#pragma unroll
        for (int e = 1; e < EE; ++e) if (l[e] > l[i0]) i0 = e;
        int i1 = (i0 == 0) ? 1 : 0;
#pragma unroll
        for (int e = 0; e < EE; ++e) if (e != i0 && l[e] > l[i1]) i1 = e;
        const float e1 = __expf(l[i1] - l[i0]);
        const float s = 1.f + e1;
        const float w0 = 1.f / s;
        const float w1 = e1 / s;
        float* wr = wout + (size_t)t * EE;
#pragma unroll
        for (int e = 0; e < EE; ++e) wr[e] = 0.f;
        wr[i0] = w0;
        wr[i1] = w1;
        const int p0 = atomicAdd(&counts[i0], 1);
        lists[i0 * TT + p0] = t; gatew[i0 * TT + p0] = w0;
        const int p1 = atomicAdd(&counts[i1], 1);
        lists[i1 * TT + p1] = t; gatew[i1 * TT + p1] = w1;
    }
}

// ---------------------------------------------------------------------------
// Kernel 2: grouped expert SwiGLU with WMMA f16.
// grid = (T/16 tiles, E experts), 256 threads = 8 wave32.
// ---------------------------------------------------------------------------
__global__ __launch_bounds__(256) void moe_expert_kernel(
    const float* __restrict__ W1a, const float* __restrict__ b1a,
    const float* __restrict__ W1b, const float* __restrict__ b1b,
    const float* __restrict__ W2,  const float* __restrict__ b2,
    const half_t* __restrict__ xn, const int* __restrict__ counts,
    const int* __restrict__ lists, const float* __restrict__ gatew,
    float* __restrict__ out)
{
    __shared__ half_t sXn[16][DD];   // 32 KB  gathered activations
    __shared__ half_t sH[16][FF];    // 64 KB  hidden SwiGLU activations
    __shared__ half_t sWa[128][40];  // 10 KB  weight tile, transposed [n][k], padded stride
    __shared__ half_t sWb[128][40];  // 10 KB
    __shared__ int   sRow[16];
    __shared__ float sG[16];

    const int e = blockIdx.y;
    const int tile = blockIdx.x;
    const int cnt = counts[e];
    if (tile * 16 >= cnt) return;    // uniform early-exit, no barriers skipped

    const int tid = threadIdx.x;
    if (tid < 16) {
        const int idx = tile * 16 + tid;
        if (idx < cnt) { sRow[tid] = lists[e * TT + idx]; sG[tid] = gatew[e * TT + idx]; }
        else           { sRow[tid] = lists[e * TT];       sG[tid] = 0.0f; } // pad row, zero gate
    }
    __syncthreads();

    // Gather the 16 token rows (f16) into LDS: 16 threads/row, 128B chunks.
    // Uses gfx1250 async global->LDS DMA (ASYNCcnt) when available.
    {
        const int r = tid >> 4;
        const int j = tid & 15;
        const char* src = (const char*)(xn + (size_t)sRow[r] * DD + j * 64);
        char* dst = (char*)(&sXn[r][j * 64]);
#ifdef USE_ASYNC_LDS
        gv4i_t* gs = (gv4i_t*)(uintptr_t)src;
        lv4i_t* ls = (lv4i_t*)(unsigned)(uintptr_t)dst;
#pragma unroll
        for (int q = 0; q < 8; ++q)
            __builtin_amdgcn_global_load_async_to_lds_b128(gs + q, ls + q, 0, 0);
        __builtin_amdgcn_s_wait_asynccnt(0);
#else
        const uint4* s4 = (const uint4*)src;
        uint4* d4 = (uint4*)dst;
#pragma unroll
        for (int q = 0; q < 8; ++q) d4[q] = s4[q];
#endif
    }
    __syncthreads();

    const int wave = tid >> 5;       // 0..7, each owns a 16-wide N slice
    const int lane = tid & 31;
    const int halfsel = lane >> 4;   // lane group selects K/M halves per WMMA layout
    const int lm = lane & 15;

    // ---- Phase 1: h[16][2048] = silu(x@W1a+b1a) * (x@W1b+b1b) ----
    for (int f0 = 0; f0 < FF; f0 += 128) {
        floatx8 accA = {};
        floatx8 accB = {};
        for (int k0 = 0; k0 < DD; k0 += 32) {
            __syncthreads(); // previous tile fully consumed
            const float4* ga4 = (const float4*)(W1a + ((size_t)e * DD + k0) * FF + f0);
            const float4* gb4 = (const float4*)(W1b + ((size_t)e * DD + k0) * FF + f0);
#pragma unroll
            for (int it = 0; it < 4; ++it) {
                const int li = tid + it * 256;  // covers (k:32) x (n4:32)
                const int k = li >> 5;
                const int n4 = li & 31;
                const float4 wa = ga4[k * (FF / 4) + n4];
                const float4 wb = gb4[k * (FF / 4) + n4];
                const int n = n4 * 4;
                sWa[n + 0][k] = (half_t)wa.x; sWa[n + 1][k] = (half_t)wa.y;
                sWa[n + 2][k] = (half_t)wa.z; sWa[n + 3][k] = (half_t)wa.w;
                sWb[n + 0][k] = (half_t)wb.x; sWb[n + 1][k] = (half_t)wb.y;
                sWb[n + 2][k] = (half_t)wb.z; sWb[n + 3][k] = (half_t)wb.w;
            }
            if (k0 + 32 < DD) {
                __builtin_prefetch((const float*)ga4 + (size_t)32 * FF + tid * 4, 0, 1);
                __builtin_prefetch((const float*)gb4 + (size_t)32 * FF + tid * 4, 0, 1);
            }
            __syncthreads();

            const halfx16 af = ld_frag(&sXn[lm][k0 + halfsel * 8],
                                       &sXn[lm][k0 + 16 + halfsel * 8]);
            const int c = wave * 16 + lm;
            const halfx16 ba = ld_frag(&sWa[c][halfsel * 8], &sWa[c][16 + halfsel * 8]);
            const halfx16 bb = ld_frag(&sWb[c][halfsel * 8], &sWb[c][16 + halfsel * 8]);
            accA = __builtin_amdgcn_wmma_f32_16x16x32_f16(false, af, false, ba,
                                                          (short)0, accA, false, false);
            accB = __builtin_amdgcn_wmma_f32_16x16x32_f16(false, af, false, bb,
                                                          (short)0, accB, false, false);
        }
        const int fcol = f0 + wave * 16 + lm;
        const float biasA = b1a[e * FF + fcol];
        const float biasB = b1b[e * FF + fcol];
#pragma unroll
        for (int r = 0; r < 8; ++r) {
            const int m = r + 8 * halfsel;
            const float a = accA[r] + biasA;
            const float b = accB[r] + biasB;
            const float hval = (a / (1.0f + __expf(-a))) * b; // silu(a)*b
            sH[m][fcol] = (half_t)hval;
        }
    }
    __syncthreads();

    // ---- Phase 2: y = h @ W2 + b2, gate-scaled atomic accumulate ----
    for (int n0 = 0; n0 < DD; n0 += 128) {
        floatx8 acc = {};
        for (int k0 = 0; k0 < FF; k0 += 32) {
            __syncthreads();
            const float4* gw4 = (const float4*)(W2 + ((size_t)e * FF + k0) * DD + n0);
#pragma unroll
            for (int it = 0; it < 4; ++it) {
                const int li = tid + it * 256;
                const int k = li >> 5;
                const int n4 = li & 31;
                const float4 w = gw4[k * (DD / 4) + n4];
                const int n = n4 * 4;
                sWa[n + 0][k] = (half_t)w.x; sWa[n + 1][k] = (half_t)w.y;
                sWa[n + 2][k] = (half_t)w.z; sWa[n + 3][k] = (half_t)w.w;
            }
            if (k0 + 32 < FF)
                __builtin_prefetch((const float*)gw4 + (size_t)32 * DD + tid * 4, 0, 1);
            __syncthreads();

            const halfx16 af = ld_frag(&sH[lm][k0 + halfsel * 8],
                                       &sH[lm][k0 + 16 + halfsel * 8]);
            const int c = wave * 16 + lm;
            const halfx16 bf = ld_frag(&sWa[c][halfsel * 8], &sWa[c][16 + halfsel * 8]);
            acc = __builtin_amdgcn_wmma_f32_16x16x32_f16(false, af, false, bf,
                                                         (short)0, acc, false, false);
        }
        const int ncol = n0 + wave * 16 + lm;
        const float bias = b2[e * DD + ncol];
#pragma unroll
        for (int r = 0; r < 8; ++r) {
            const int m = r + 8 * halfsel;
            const float val = (acc[r] + bias) * sG[m];
            atomicAdd(&out[(size_t)sRow[m] * DD + ncol], val);
        }
    }
}

// ---------------------------------------------------------------------------
extern "C" void kernel_launch(void* const* d_in, const int* in_sizes, int n_in,
                              void* d_out, int out_size, void* d_ws, size_t ws_size,
                              hipStream_t stream) {
    (void)in_sizes; (void)n_in; (void)out_size; (void)ws_size;
    const float* x    = (const float*)d_in[0];
    const float* rmsw = (const float*)d_in[1];
    const float* Wg   = (const float*)d_in[2];
    const float* bg   = (const float*)d_in[3];
    const float* W1a  = (const float*)d_in[4];
    const float* b1a  = (const float*)d_in[5];
    const float* W1b  = (const float*)d_in[6];
    const float* b1b  = (const float*)d_in[7];
    const float* W2   = (const float*)d_in[8];
    const float* b2   = (const float*)d_in[9];

    float* out  = (float*)d_out;                 // [B,S,D]
    float* wout = out + (size_t)TT * DD;         // [B,S,E]

    char* ws = (char*)d_ws;
    half_t* xn = (half_t*)ws;                    // [T,D] f16 normalized activations
    size_t off = (size_t)TT * DD * sizeof(half_t);
    int* counts = (int*)(ws + off);              off += 256;
    int* lists  = (int*)(ws + off);              off += (size_t)EE * TT * sizeof(int);
    float* gatew = (float*)(ws + off);

    (void)hipMemsetAsync(counts, 0, EE * sizeof(int), stream);
    rmsnorm_gate_kernel<<<TT, 256, 0, stream>>>(x, rmsw, Wg, bg, xn, counts,
                                                lists, gatew, wout, out);
    dim3 grid(TT / 16, EE);
    moe_expert_kernel<<<grid, 256, 0, stream>>>(W1a, b1a, W1b, b1b, W2, b2,
                                                xn, counts, lists, gatew, out);
}
